// NeuralCE_Heisenberg_Lite_28149215658681
// MI455X (gfx1250) — matrix operand order repaired and checked by reading, not compiled, not run on hardware
//
#include <hip/hip_runtime.h>
#include <hip/hip_bf16.h>
#include <math.h>

typedef __attribute__((ext_vector_type(16))) _Float16 v16h;
typedef __attribute__((ext_vector_type(8)))  float    v8f;

#define NATOM 32768
#define MNBR  32
#define FDIM  64

// Branchless fast softplus: max(x,0) + log(1+exp(-|x|)).
// Uses hardware v_exp_f32 / v_log_f32 (TRANS, co-issues with XDL WMMA).
__device__ __forceinline__ float softplusf_(float x) {
  return fmaxf(x, 0.0f) + __logf(1.0f + __expf(-fabsf(x)));
}
// Hardware v_rcp_f32 instead of IEEE division (v_div_scale + Newton chain).
__device__ __forceinline__ float sigmoidf_(float x) {
  return __builtin_amdgcn_rcpf(1.0f + __expf(-x));
}

// ---------------------------------------------------------------------------
// Generic WMMA GEMM: C[rows x NCOL] = act(A[rows x KDIM] @ B[KDIM x NCOL] + bias)
// A,B,C fp32 row-major; compute in f16 WMMA with f32 accumulate.
// Block = 128 threads = 4 waves; each wave owns one 16-row tile.
// ACT: 0 = none, 1 = softplus
// ---------------------------------------------------------------------------
template<int KDIM, int NCOL, int ACT>
__global__ __launch_bounds__(128) void gemm_wmma_kernel(
    const float* __restrict__ A, const float* __restrict__ B,
    const float* __restrict__ bias, float* __restrict__ C, int rows)
{
  constexpr int KPAD  = (KDIM + 31) & ~31;
  constexpr int NFRAG = KPAD / 32;
  constexpr int NTILE = NCOL / 16;

  __shared__ _Float16 Bs[KPAD * NCOL];
  __shared__ float biasS[NCOL];

  const int tid = threadIdx.x;
  for (int i = tid; i < KPAD * NCOL; i += 128) {
    int k = i / NCOL, n = i % NCOL;
    Bs[i] = (k < KDIM) ? (_Float16)B[(size_t)k * NCOL + n] : (_Float16)0.0f;
  }
  for (int i = tid; i < NCOL; i += 128) biasS[i] = bias ? bias[i] : 0.0f;
  __syncthreads();

  const int wave = tid >> 5, lane = tid & 31;
  const int tile = blockIdx.x * 4 + wave;
  const int row0 = tile * 16;
  if (row0 >= rows) return;

  const int r   = lane & 15;      // A row within tile, C column
  const int kh  = lane >> 4;      // K-half selector
  const int col = lane & 15;
  const size_t grow = (size_t)(row0 + r);

  // A fragments: lane holds row r; elems 0..7 -> K = kf*32 + kh*8 + i,
  // elems 8..15 -> K = kf*32 + 16 + kh*8 + (i-8)
  v16h afrag[NFRAG];
  #pragma unroll
  for (int kf = 0; kf < NFRAG; ++kf) {
    #pragma unroll
    for (int i = 0; i < 16; ++i) {
      int k = kf * 32 + (i < 8 ? kh * 8 + i : 16 + kh * 8 + (i - 8));
      float v = (k < KDIM) ? A[grow * KDIM + k] : 0.0f;
      afrag[kf][i] = (_Float16)v;
    }
  }

  #pragma unroll
  for (int n = 0; n < NTILE; ++n) {
    // B fragment: lane holds col; elems 0..7 -> K = kf*32 + kh*8 + i, elems 8..15 -> K = kf*32 + 16 + kh*8 + (i-8)
    v16h bfrag[NFRAG];
    #pragma unroll
    for (int kf = 0; kf < NFRAG; ++kf) {
      #pragma unroll
      for (int i = 0; i < 16; ++i)
        bfrag[kf][i] = Bs[(kf * 32 + (i < 8 ? kh * 8 + i : 16 + kh * 8 + (i - 8))) * NCOL + n * 16 + col];
    }
    float bv = biasS[n * 16 + col];
    v8f c;
    #pragma unroll
    for (int i = 0; i < 8; ++i) c[i] = bv;
    #pragma unroll
    for (int kf = 0; kf < NFRAG; ++kf)
      c = __builtin_amdgcn_wmma_f32_16x16x32_f16(false, afrag[kf], false,
                                                 bfrag[kf], (short)0, c,
                                                 false, false);
    #pragma unroll
    for (int i = 0; i < 8; ++i) {
      float v = c[i];
      if (ACT == 1) v = softplusf_(v);
      C[(size_t)(row0 + i + 8 * kh) * NCOL + n * 16 + col] = v;
    }
  }
}

// ---------------------------------------------------------------------------
// nbr_h = nbr_fea @ W_nbrh + b_nbrh   ([rows,8] @ [8,8]), one thread per row
// ---------------------------------------------------------------------------
__global__ void nbrh_kernel(const float* __restrict__ nbr_fea,
                            const float* __restrict__ W,
                            const float* __restrict__ b,
                            float* __restrict__ out, int rows)
{
  int rIdx = blockIdx.x * blockDim.x + threadIdx.x;
  if (rIdx >= rows) return;
  float in[8];
  #pragma unroll
  for (int t = 0; t < 8; ++t) in[t] = nbr_fea[(size_t)rIdx * 8 + t];
  #pragma unroll
  for (int j = 0; j < 8; ++j) {
    float acc = b[j];
    #pragma unroll
    for (int t = 0; t < 8; ++t) acc += in[t] * W[t * 8 + j];
    out[(size_t)rIdx * 8 + j] = acc;
  }
}

// ---------------------------------------------------------------------------
// LayerNorm over F=64, one thread per atom
// ---------------------------------------------------------------------------
__global__ void ln_kernel(const float* __restrict__ x,
                          const float* __restrict__ scale,
                          const float* __restrict__ bias,
                          float* __restrict__ y, int n)
{
  int a = blockIdx.x * blockDim.x + threadIdx.x;
  if (a >= n) return;
  const float* row = x + (size_t)a * FDIM;
  float mu = 0.0f;
  #pragma unroll
  for (int k = 0; k < FDIM; ++k) mu += row[k];
  mu *= (1.0f / FDIM);
  float var = 0.0f;
  #pragma unroll
  for (int k = 0; k < FDIM; ++k) { float d = row[k] - mu; var += d * d; }
  var *= (1.0f / FDIM);
  float inv = rsqrtf(var + 1e-6f);
  #pragma unroll
  for (int k = 0; k < FDIM; ++k)
    y[(size_t)a * FDIM + k] = (row[k] - mu) * inv * scale[k] + bias[k];
}

// ---------------------------------------------------------------------------
// Fused conv: for each 16-atom tile (one wave), loop over 32 neighbors:
//   inter = phi_c * phi_n[idx] * (nbr_h @ We + be)   (built in A-fragment regs)
//   acc  += sigmoid(inter@Wg+bg) * softplus(inter@Wm+bm)   (16 WMMAs / nbr)
// atom_out = atom_in + acc.
// Wg/Wm are pre-swizzled into *fragment-layout LDS*. An opaque per-iteration
// offset (empty inline asm) makes the fragment addresses loop-variant so
// LICM cannot hoist the 16 B-fragment loads into long-lived registers
// (which previously caused either VGPR-MSB thrash or scratch spills).
// Each fragment is one v16h = 2x ds_load_b128 per use, overlapped with the
// XDL WMMAs; live set stays well under 256 VGPRs.
// ---------------------------------------------------------------------------
__global__ __launch_bounds__(128) void conv_fused_kernel(
    const float* __restrict__ atom_in, const float* __restrict__ phiC,
    const float* __restrict__ phiN,    const float* __restrict__ nbrH,
    const int*   __restrict__ idx,
    const float* __restrict__ We, const float* __restrict__ be,
    const float* __restrict__ Wg, const float* __restrict__ bg,
    const float* __restrict__ Wm, const float* __restrict__ bm,
    float* __restrict__ atom_out, int natoms)
{
  // fragment layout: [(n*2+kf)*32 + lane]*16 + i   (32 bytes per lane-slot)
  __shared__ __align__(32) _Float16 fragG[8 * 32 * 16];
  __shared__ __align__(32) _Float16 fragM[8 * 32 * 16];
  __shared__ float WeS[8 * FDIM];
  __shared__ float beS[FDIM], bgS[FDIM], bmS[FDIM];

  const int tid  = threadIdx.x;
  const int wave = tid >> 5, lane = tid & 31;
  const int kh   = lane >> 4;
  const int col  = lane & 15;

  for (int i = tid; i < 8 * FDIM; i += 128) WeS[i] = We[i];
  for (int i = tid; i < FDIM; i += 128) {
    beS[i] = be[i]; bgS[i] = bg[i]; bmS[i] = bm[i];
  }
  // wave w builds the fragments of n-tile w (B layout: lane=col, K=kf*32+(i<8 ? kh*8+i : 16+kh*8+(i-8)))
  {
    const int n = wave;
    #pragma unroll
    for (int kf = 0; kf < 2; ++kf) {
      #pragma unroll
      for (int i = 0; i < 16; ++i) {
        const int kk = kf * 32 + (i < 8 ? kh * 8 + i : 16 + kh * 8 + (i - 8));
        const int slot = ((n * 2 + kf) * 32 + lane) * 16 + i;
        fragG[slot] = (_Float16)Wg[kk * FDIM + n * 16 + col];
        fragM[slot] = (_Float16)Wm[kk * FDIM + n * 16 + col];
      }
    }
  }
  __syncthreads();

  const int tile = blockIdx.x * 4 + wave;
  const int row0 = tile * 16;
  if (row0 >= natoms) return;

  const int r = lane & 15;
  const size_t grow = (size_t)(row0 + r);

  // K-run bases for this lane's A-fragment positions (4 runs of 8 consecutive K)
  int kb[4];
  kb[0] = kh * 8;       kb[1] = 16 + kh * 8;
  kb[2] = 32 + kh * 8;  kb[3] = 48 + kh * 8;

  // phi_c for this lane's row, in fragment order
  float pc[4][8];
  #pragma unroll
  for (int run = 0; run < 4; ++run) {
    const float4 lo = *(const float4*)(phiC + grow * FDIM + kb[run]);
    const float4 hi = *(const float4*)(phiC + grow * FDIM + kb[run] + 4);
    pc[run][0] = lo.x; pc[run][1] = lo.y; pc[run][2] = lo.z; pc[run][3] = lo.w;
    pc[run][4] = hi.x; pc[run][5] = hi.y; pc[run][6] = hi.z; pc[run][7] = hi.w;
  }

  v8f acc[4];
  #pragma unroll
  for (int n = 0; n < 4; ++n)
    #pragma unroll
    for (int i = 0; i < 8; ++i) acc[n][i] = 0.0f;

  // Opaque offset: redefined every iteration by empty asm so the B-fragment
  // LDS loads below are not loop-invariant and cannot be hoisted/spilled.
  int fragOff = 0;

  #pragma unroll 1
  for (int m = 0; m < MNBR; ++m) {
    asm volatile("" : "+v"(fragOff));
    const int j = idx[grow * MNBR + m];

    // gather phi_n row (8 MB table -> L2-resident)
    float pn[4][8];
    const float* pnrow = phiN + (size_t)j * FDIM;
    #pragma unroll
    for (int run = 0; run < 4; ++run) {
      const float4 lo = *(const float4*)(pnrow + kb[run]);
      const float4 hi = *(const float4*)(pnrow + kb[run] + 4);
      pn[run][0] = lo.x; pn[run][1] = lo.y; pn[run][2] = lo.z; pn[run][3] = lo.w;
      pn[run][4] = hi.x; pn[run][5] = hi.y; pn[run][6] = hi.z; pn[run][7] = hi.w;
    }

    // neighbor edge vector (8)
    float nb[8];
    const float4 n0 = *(const float4*)(nbrH + (grow * MNBR + m) * 8);
    const float4 n1 = *(const float4*)(nbrH + (grow * MNBR + m) * 8 + 4);
    nb[0] = n0.x; nb[1] = n0.y; nb[2] = n0.z; nb[3] = n0.w;
    nb[4] = n1.x; nb[5] = n1.y; nb[6] = n1.z; nb[7] = n1.w;

    // phi_e = nb @ We + be  (VALU; co-executes with WMMA)
    float pe[4][8];
    #pragma unroll
    for (int run = 0; run < 4; ++run) {
      #pragma unroll
      for (int i = 0; i < 8; ++i) pe[run][i] = beS[kb[run] + i];
      #pragma unroll
      for (int t = 0; t < 8; ++t) {
        const float nv = nb[t];
        #pragma unroll
        for (int i = 0; i < 8; ++i)
          pe[run][i] += nv * WeS[t * FDIM + kb[run] + i];
      }
    }

    // inter tile in A-fragment layout
    v16h a0, a1;
    #pragma unroll
    for (int i = 0; i < 8; ++i) {
      a0[i]     = (_Float16)(pc[0][i] * pn[0][i] * pe[0][i]);
      a0[i + 8] = (_Float16)(pc[1][i] * pn[1][i] * pe[1][i]);
      a1[i]     = (_Float16)(pc[2][i] * pn[2][i] * pe[2][i]);
      a1[i + 8] = (_Float16)(pc[3][i] * pn[3][i] * pe[3][i]);
    }

    #pragma unroll
    for (int n = 0; n < 4; ++n) {
      const v16h bg0 = *(const v16h*)&fragG[((n * 2 + 0) * 32 + lane) * 16 + fragOff];
      const v16h bg1 = *(const v16h*)&fragG[((n * 2 + 1) * 32 + lane) * 16 + fragOff];
      const v16h bm0 = *(const v16h*)&fragM[((n * 2 + 0) * 32 + lane) * 16 + fragOff];
      const v16h bm1 = *(const v16h*)&fragM[((n * 2 + 1) * 32 + lane) * 16 + fragOff];
      v8f cg, cm;
      const float bgv = bgS[n * 16 + col];
      const float bmv = bmS[n * 16 + col];
      #pragma unroll
      for (int i = 0; i < 8; ++i) { cg[i] = bgv; cm[i] = bmv; }
      cg = __builtin_amdgcn_wmma_f32_16x16x32_f16(false, a0, false, bg0, (short)0, cg, false, false);
      cg = __builtin_amdgcn_wmma_f32_16x16x32_f16(false, a1, false, bg1, (short)0, cg, false, false);
      cm = __builtin_amdgcn_wmma_f32_16x16x32_f16(false, a0, false, bm0, (short)0, cm, false, false);
      cm = __builtin_amdgcn_wmma_f32_16x16x32_f16(false, a1, false, bm1, (short)0, cm, false, false);
      #pragma unroll
      for (int i = 0; i < 8; ++i)
        acc[n][i] += sigmoidf_(cg[i]) * softplusf_(cm[i]);
    }
  }

  // residual + store (C layout: row = row0 + vgpr + 8*kh, col = n*16 + lane%16)
  #pragma unroll
  for (int n = 0; n < 4; ++n) {
    #pragma unroll
    for (int i = 0; i < 8; ++i) {
      const size_t orow = (size_t)(row0 + i + 8 * kh);
      const int    ocol = n * 16 + col;
      atom_out[orow * FDIM + ocol] = atom_in[orow * FDIM + ocol] + acc[n][i];
    }
  }
}

// ---------------------------------------------------------------------------
// Final per-atom energy: E_chem (h2 @ W_r3 + b_r3) + E_mag (J-coupling).
// Writes per-block partial sums (deterministic; no float atomics).
// ---------------------------------------------------------------------------
__global__ __launch_bounds__(256) void final_kernel(
    const float* __restrict__ h2,  const float* __restrict__ Wr3, const float* __restrict__ br3,
    const float* __restrict__ pi,  const float* __restrict__ pj,
    const float* __restrict__ Wne, const float* __restrict__ bj1,
    const float* __restrict__ Wj2, const float* __restrict__ bj2,
    const float* __restrict__ nbrH, const float* __restrict__ nbr_fea,
    const int*   __restrict__ idx,  const float* __restrict__ spins,
    float* __restrict__ partials, int n)
{
  __shared__ float red[256];
  const int a = blockIdx.x * 256 + threadIdx.x;
  float esite = 0.0f;
  if (a < n) {
    float echem = br3[0];
    #pragma unroll
    for (int k = 0; k < FDIM; ++k) echem += h2[(size_t)a * FDIM + k] * Wr3[k];

    float pia[16];
    #pragma unroll
    for (int k = 0; k < 16; ++k) pia[k] = pi[(size_t)a * 16 + k];
    const float sa = spins[a];

    float emag = 0.0f;
    for (int m = 0; m < MNBR; ++m) {
      const size_t em = (size_t)a * MNBR + m;
      bool any = false;
      #pragma unroll
      for (int t = 0; t < 8; ++t) any |= (nbr_fea[em * 8 + t] > 0.0f);
      if (!any) continue;
      const int j = idx[em];
      float nb[8];
      #pragma unroll
      for (int t = 0; t < 8; ++t) nb[t] = nbrH[em * 8 + t];
      float jij = bj2[0];
      #pragma unroll
      for (int k = 0; k < 16; ++k) {
        float ne = bj1[k];
        #pragma unroll
        for (int t = 0; t < 8; ++t) ne += nb[t] * Wne[t * 16 + k];
        const float jh = softplusf_(pia[k] + pj[(size_t)j * 16 + k] + ne);
        jij += jh * Wj2[k];
      }
      emag += jij * sa * spins[j];
    }
    esite = echem + emag;
  }
  red[threadIdx.x] = esite;
  __syncthreads();
  for (int s = 128; s > 0; s >>= 1) {
    if (threadIdx.x < s) red[threadIdx.x] += red[threadIdx.x + s];
    __syncthreads();
  }
  if (threadIdx.x == 0) partials[blockIdx.x] = red[0];
}

__global__ void reduce_kernel(const float* __restrict__ partials, int n,
                              float* __restrict__ out)
{
  __shared__ float red[256];
  float v = 0.0f;
  for (int i = threadIdx.x; i < n; i += 256) v += partials[i];
  red[threadIdx.x] = v;
  __syncthreads();
  for (int s = 128; s > 0; s >>= 1) {
    if (threadIdx.x < s) red[threadIdx.x] += red[threadIdx.x + s];
    __syncthreads();
  }
  if (threadIdx.x == 0) out[0] = red[0];
}

// ---------------------------------------------------------------------------
extern "C" void kernel_launch(void* const* d_in, const int* in_sizes, int n_in,
                              void* d_out, int out_size, void* d_ws, size_t ws_size,
                              hipStream_t stream) {
  (void)in_sizes; (void)n_in; (void)out_size; (void)ws_size;

  const float* atom_fea = (const float*)d_in[0];
  const float* nbr_fea  = (const float*)d_in[1];
  const int*   nbr_idx  = (const int*)  d_in[2];
  const float* spins    = (const float*)d_in[3];
  const float* W_embed  = (const float*)d_in[4];
  const float* b_embed  = (const float*)d_in[5];
  const float* W_nbrh   = (const float*)d_in[6];
  const float* b_nbrh   = (const float*)d_in[7];
  const float* ln_scale = (const float*)d_in[8];
  const float* ln_bias  = (const float*)d_in[9];
  const float* Wc       = (const float*)d_in[10];
  const float* bc       = (const float*)d_in[11];
  const float* Wn       = (const float*)d_in[12];
  const float* bn       = (const float*)d_in[13];
  const float* We       = (const float*)d_in[14];
  const float* be       = (const float*)d_in[15];
  const float* Wg       = (const float*)d_in[16];
  const float* bg       = (const float*)d_in[17];
  const float* Wm       = (const float*)d_in[18];
  const float* bm       = (const float*)d_in[19];
  const float* W_r1     = (const float*)d_in[20];
  const float* b_r1     = (const float*)d_in[21];
  const float* W_r2     = (const float*)d_in[22];
  const float* b_r2     = (const float*)d_in[23];
  const float* W_r3     = (const float*)d_in[24];
  const float* b_r3     = (const float*)d_in[25];
  const float* W_j1     = (const float*)d_in[26];
  const float* b_j1     = (const float*)d_in[27];
  const float* W_j2     = (const float*)d_in[28];
  const float* b_j2     = (const float*)d_in[29];

  const int N = NATOM;
  const int NM = NATOM * MNBR;

  // workspace carve-out (fp32)
  float* ws = (float*)d_ws;
  size_t off = 0;
  float* atomA   = ws + off; off += (size_t)N * 64;
  float* atomB   = ws + off; off += (size_t)N * 64;
  float* normed  = ws + off; off += (size_t)N * 64;
  float* phiC    = ws + off; off += (size_t)N * 64;
  float* phiN    = ws + off; off += (size_t)N * 64;
  float* nbrH    = ws + off; off += (size_t)NM * 8;
  float* h1      = ws + off; off += (size_t)N * 128;
  float* h2b     = ws + off; off += (size_t)N * 64;
  float* piB     = ws + off; off += (size_t)N * 16;
  float* pjB     = ws + off; off += (size_t)N * 16;
  float* parts   = ws + off; off += 256;

  const int gemmGridN = (N / 16 + 3) / 4;   // 512 blocks, 4 wave-tiles each

  // 1) embed: atom = atom_fea @ W_embed + b_embed   (K=92 -> pad 96)
  gemm_wmma_kernel<92, 64, 0><<<gemmGridN, 128, 0, stream>>>(
      atom_fea, W_embed, b_embed, atomA, N);

  // 2) nbr_h = nbr_fea @ W_nbrh + b_nbrh
  nbrh_kernel<<<(NM + 255) / 256, 256, 0, stream>>>(nbr_fea, W_nbrh, b_nbrh, nbrH, NM);

  // 3) two conv layers (ping-pong atomA <-> atomB)
  for (int c = 0; c < 2; ++c) {
    const float* ain  = (c == 0) ? atomA : atomB;
    float*       aout = (c == 0) ? atomB : atomA;
    ln_kernel<<<(N + 255) / 256, 256, 0, stream>>>(
        ain, ln_scale + c * 64, ln_bias + c * 64, normed, N);
    gemm_wmma_kernel<64, 64, 0><<<gemmGridN, 128, 0, stream>>>(
        normed, Wc + c * 4096, bc + c * 64, phiC, N);
    gemm_wmma_kernel<64, 64, 0><<<gemmGridN, 128, 0, stream>>>(
        normed, Wn + c * 4096, bn + c * 64, phiN, N);
    conv_fused_kernel<<<gemmGridN, 128, 0, stream>>>(
        ain, phiC, phiN, nbrH, nbr_idx,
        We + c * 512, be + c * 64,
        Wg + c * 4096, bg + c * 64,
        Wm + c * 4096, bm + c * 64,
        aout, N);
  }
  // after two convs the result is in atomA

  // 4) readout MLP with fused softplus
  gemm_wmma_kernel<64, 128, 1><<<gemmGridN, 128, 0, stream>>>(
      atomA, W_r1, b_r1, h1, N);
  gemm_wmma_kernel<128, 64, 1><<<gemmGridN, 128, 0, stream>>>(
      h1, W_r2, b_r2, h2b, N);

  // 5) J-coupling projections: p_i = atom @ Wa, p_j = atom @ Wb (rows 0..63 / 64..127 of W_j1)
  gemm_wmma_kernel<64, 16, 0><<<gemmGridN, 128, 0, stream>>>(
      atomA, W_j1, (const float*)nullptr, piB, N);
  gemm_wmma_kernel<64, 16, 0><<<gemmGridN, 128, 0, stream>>>(
      atomA, W_j1 + 64 * 16, (const float*)nullptr, pjB, N);

  // 6) per-atom energies + deterministic two-stage sum
  const int finalBlocks = (N + 255) / 256;   // 128
  final_kernel<<<finalBlocks, 256, 0, stream>>>(
      h2b, W_r3, b_r3, piB, pjB, W_j1 + 128 * 16, b_j1, W_j2, b_j2,
      nbrH, nbr_fea, nbr_idx, spins, parts, N);
  reduce_kernel<<<1, 256, 0, stream>>>(parts, finalBlocks, (float*)d_out);
}
